// MoE_48962627174702
// MI455X (gfx1250) — matrix operand order repaired and checked by reading, compiled, not run on hardware
//
#include <hip/hip_runtime.h>

// ---------------- problem constants ----------------
constexpr int TOKS = 4096;          // B*S
constexpr int DIMX = 1024;
constexpr int HIDX = 2816;
constexpr int NEXP = 8;
constexpr int PAIRS = TOKS * 2;     // top-2 routing
constexpr int BM = 128, BN = 128, BK = 64;
constexpr int LDA = BK + 8;         // LDS row stride in bf16 elems (16B aligned)
constexpr int TILE = BM * LDA;      // elems per LDS tile

typedef __attribute__((ext_vector_type(16))) __bf16 v16bf;
typedef __attribute__((ext_vector_type(8)))  __bf16 v8bf;
typedef __attribute__((ext_vector_type(8)))  float  v8f;

typedef __attribute__((address_space(3))) unsigned short lds_u16;

__device__ __forceinline__ unsigned short f2bf(float f) {
    unsigned u = __float_as_uint(f);
    unsigned r = u + 0x7FFFu + ((u >> 16) & 1u);   // RNE
    return (unsigned short)(r >> 16);
}

__device__ __forceinline__ v16bf frag16(const unsigned short* p0, const unsigned short* p1) {
    v8bf a = *(const v8bf*)p0;
    v8bf b = *(const v8bf*)p1;
    return __builtin_shufflevector(a, b, 0,1,2,3,4,5,6,7,8,9,10,11,12,13,14,15);
}

// LDS byte offset of a generic pointer known to point into LDS
__device__ __forceinline__ unsigned lds_off(unsigned short* p) {
    return (unsigned)(unsigned long long)(lds_u16*)p;
}

// async DMA: 16 bytes global -> LDS, tracked by ASYNCcnt (CDNA5)
__device__ __forceinline__ void async_cp16(unsigned lds_addr, const unsigned short* g) {
    asm volatile("global_load_async_to_lds_b128 %0, %1, off"
                 :: "v"(lds_addr), "v"((unsigned long long)(size_t)g)
                 : "memory");
}

// ---------------- utility kernels ----------------
__global__ void __launch_bounds__(256) zero_kernel(float* out, int* cnt) {
    int i = blockIdx.x * 256 + threadIdx.x;
    out[i] = 0.0f;
    if (i < NEXP) cnt[i] = 0;
}

__global__ void __launch_bounds__(256) cvt_kernel(const float* __restrict__ src,
                                                  unsigned short* __restrict__ dst, int n4) {
    int i = blockIdx.x * 256 + threadIdx.x;
    if (i >= n4) return;
    float4 v = ((const float4*)src)[i];
    ushort4 o;
    o.x = f2bf(v.x); o.y = f2bf(v.y); o.z = f2bf(v.z); o.w = f2bf(v.w);
    ((ushort4*)dst)[i] = o;
}

// one wave (32 lanes) per token: 8 dot products, wave reduce, top-2 + softmax
__global__ void __launch_bounds__(256) gating_kernel(const float* __restrict__ x,
                                                     const float* __restrict__ Wg,
                                                     int* __restrict__ tkid,
                                                     float* __restrict__ tkw,
                                                     int* __restrict__ cnt) {
    int wave = threadIdx.x >> 5, lane = threadIdx.x & 31;
    int t = blockIdx.x * 8 + wave;
    if (t >= TOKS) return;
    const float* xr = x + (size_t)t * DIMX;
    float s[NEXP];
    #pragma unroll
    for (int e = 0; e < NEXP; ++e) s[e] = 0.0f;
    for (int d = lane; d < DIMX; d += 32) {
        float xv = xr[d];
        #pragma unroll
        for (int e = 0; e < NEXP; ++e) s[e] += xv * Wg[e * DIMX + d];
    }
    #pragma unroll
    for (int e = 0; e < NEXP; ++e)
        #pragma unroll
        for (int off = 16; off > 0; off >>= 1)
            s[e] += __shfl_xor(s[e], off, 32);
    if (lane == 0) {
        int i1 = 0; float v1 = s[0];
        #pragma unroll
        for (int e = 1; e < NEXP; ++e) if (s[e] > v1) { v1 = s[e]; i1 = e; }
        int i2 = -1; float v2 = -3.0e38f;
        #pragma unroll
        for (int e = 0; e < NEXP; ++e) if (e != i1 && s[e] > v2) { v2 = s[e]; i2 = e; }
        float w1 = 1.0f / (1.0f + __expf(v2 - v1));
        tkid[t * 2 + 0] = i1; tkw[t * 2 + 0] = w1;
        tkid[t * 2 + 1] = i2; tkw[t * 2 + 1] = 1.0f - w1;
        atomicAdd(&cnt[i1], 1);
        atomicAdd(&cnt[i2], 1);
    }
}

__global__ void scan_kernel(const int* __restrict__ cnt, int* __restrict__ offs,
                            int* __restrict__ cur) {
    if (threadIdx.x == 0) {
        int acc = 0;
        for (int e = 0; e < NEXP; ++e) { offs[e] = acc; cur[e] = acc; acc += cnt[e]; }
        offs[NEXP] = acc;
    }
}

__global__ void __launch_bounds__(256) scatter_kernel(const int* __restrict__ tkid,
                                                      const float* __restrict__ tkw,
                                                      int* __restrict__ cur,
                                                      int* __restrict__ ptok,
                                                      float* __restrict__ pwgt) {
    int t = blockIdx.x * 256 + threadIdx.x;
    if (t >= TOKS) return;
    #pragma unroll
    for (int s = 0; s < 2; ++s) {
        int e = tkid[t * 2 + s];
        int pos = atomicAdd(&cur[e], 1);
        ptok[pos] = t;
        pwgt[pos] = tkw[t * 2 + s];
    }
}

// ---------------- GEMM1: H = silu(X W1^T) * (X W3^T), per-expert, gathered rows ----------------
// Double-buffered LDS tiles filled by global_load_async_to_lds_b128 (ASYNCcnt).
__global__ void __launch_bounds__(256) moe_gemm1(const unsigned short* __restrict__ xb,
                                                 const unsigned short* __restrict__ W1b,
                                                 const unsigned short* __restrict__ W3b,
                                                 unsigned short* __restrict__ Hb,
                                                 const int* __restrict__ ptok,
                                                 const int* __restrict__ cnt,
                                                 const int* __restrict__ offs) {
    extern __shared__ __align__(16) unsigned short smem[];   // 2 x (A,B1,B3) tiles
    __shared__ int sTok[BM];

    constexpr int MT = TOKS / BM;
    int e = blockIdx.x / MT, mt = blockIdx.x % MT;
    int c = cnt[e];
    if (mt * BM >= c) return;
    int base = offs[e];
    int n0 = blockIdx.y * BN;

    int tid = threadIdx.x;
    int lane = tid & 31, wave = tid >> 5;
    int wm = wave >> 2, wn = wave & 3;
    int lrow = lane & 15, lhi = lane >> 4;

    for (int i = tid; i < BM; i += 256) {
        int r = mt * BM + i;
        sTok[i] = (r < c) ? ptok[base + r] : 0;
    }
    __syncthreads();

    const size_t wstride = (size_t)HIDX * DIMX;
    const unsigned short* B1 = W1b + (size_t)e * wstride + (size_t)n0 * DIMX;
    const unsigned short* B3 = W3b + (size_t)e * wstride + (size_t)n0 * DIMX;

    // issue one stage of async tile fills: 12 async instructions per thread/wave
    auto issue = [&](int buf, int k0) {
        unsigned short* dA  = smem + buf * (3 * TILE);
        unsigned short* dB1 = dA + TILE;
        unsigned short* dB3 = dB1 + TILE;
        #pragma unroll
        for (int i = tid; i < BM * (BK / 8); i += 256) {
            int r = i >> 3, ch = (i & 7) * 8;
            async_cp16(lds_off(&dA[r * LDA + ch]),
                       &xb[(size_t)sTok[r] * DIMX + k0 + ch]);
            async_cp16(lds_off(&dB1[r * LDA + ch]),
                       &B1[(size_t)r * DIMX + k0 + ch]);
            async_cp16(lds_off(&dB3[r * LDA + ch]),
                       &B3[(size_t)r * DIMX + k0 + ch]);
        }
    };

    v8f acc1[4][2], acc3[4][2];
    #pragma unroll
    for (int mi = 0; mi < 4; ++mi)
        #pragma unroll
        for (int ni = 0; ni < 2; ++ni) {
            acc1[mi][ni] = (v8f){0,0,0,0,0,0,0,0};
            acc3[mi][ni] = (v8f){0,0,0,0,0,0,0,0};
        }

    issue(0, 0);
    constexpr int NK = DIMX / BK;
    for (int it = 0; it < NK; ++it) {
        int cur = it & 1;
        if (it + 1 < NK) {
            issue(1 - cur, (it + 1) * BK);                 // prefetch next stage
            asm volatile("s_wait_asynccnt 12" ::: "memory"); // current stage landed
        } else {
            asm volatile("s_wait_asynccnt 0" ::: "memory");
        }
        __syncthreads();

        const unsigned short* cA  = smem + cur * (3 * TILE);
        const unsigned short* cB1 = cA + TILE;
        const unsigned short* cB3 = cB1 + TILE;

        #pragma unroll
        for (int kk = 0; kk < BK; kk += 32) {
            int kA0 = kk + lhi * 16, kA1 = kk + lhi * 16 + 8;
            int kB0 = kk + lhi * 16;
            v16bf af[4], b1f[2], b3f[2];
            #pragma unroll
            for (int mi = 0; mi < 4; ++mi) {
                int row = wm * 64 + mi * 16 + lrow;
                af[mi] = frag16(&cA[row * LDA + kA0], &cA[row * LDA + kA1]);
            }
            #pragma unroll
            for (int ni = 0; ni < 2; ++ni) {
                int row = wn * 32 + ni * 16 + lrow;
                b1f[ni] = frag16(&cB1[row * LDA + kB0], &cB1[row * LDA + kB0 + 8]);
                b3f[ni] = frag16(&cB3[row * LDA + kB0], &cB3[row * LDA + kB0 + 8]);
            }
            #pragma unroll
            for (int mi = 0; mi < 4; ++mi)
                #pragma unroll
                for (int ni = 0; ni < 2; ++ni) {
                    acc1[mi][ni] = __builtin_amdgcn_wmma_f32_16x16x32_bf16(
                        false, af[mi], false, b1f[ni], (short)0, acc1[mi][ni], false, false);
                    acc3[mi][ni] = __builtin_amdgcn_wmma_f32_16x16x32_bf16(
                        false, af[mi], false, b3f[ni], (short)0, acc3[mi][ni], false, false);
                }
        }
        __syncthreads();
    }

    #pragma unroll
    for (int mi = 0; mi < 4; ++mi)
        #pragma unroll
        for (int ni = 0; ni < 2; ++ni)
            #pragma unroll
            for (int j = 0; j < 8; ++j) {
                int rT = mt * BM + wm * 64 + mi * 16 + lhi * 8 + j;
                if (rT < c) {
                    int col = n0 + wn * 32 + ni * 16 + lrow;
                    float h1 = acc1[mi][ni][j], h3 = acc3[mi][ni][j];
                    float h = (h1 / (1.0f + __expf(-h1))) * h3;  // silu(h1)*h3
                    Hb[(size_t)(base + rT) * HIDX + col] = f2bf(h);
                }
            }
}

// ---------------- GEMM2: out[tok] += w * (H W2^T) ----------------
__global__ void __launch_bounds__(256) moe_gemm2(const unsigned short* __restrict__ Hb,
                                                 const unsigned short* __restrict__ W2b,
                                                 float* __restrict__ out,
                                                 const int* __restrict__ ptok,
                                                 const float* __restrict__ pwgt,
                                                 const int* __restrict__ cnt,
                                                 const int* __restrict__ offs) {
    extern __shared__ __align__(16) unsigned short smem[];   // 2 x (A,B) tiles
    __shared__ int sTok[BM];
    __shared__ float sWgt[BM];

    constexpr int MT = TOKS / BM;
    int e = blockIdx.x / MT, mt = blockIdx.x % MT;
    int c = cnt[e];
    if (mt * BM >= c) return;
    int base = offs[e];
    int n0 = blockIdx.y * BN;

    int tid = threadIdx.x;
    int lane = tid & 31, wave = tid >> 5;
    int wm = wave >> 2, wn = wave & 3;
    int lrow = lane & 15, lhi = lane >> 4;

    for (int i = tid; i < BM; i += 256) {
        int r = mt * BM + i;
        int pr = base + ((r < c) ? r : 0);
        sTok[i] = ptok[pr];
        sWgt[i] = pwgt[pr];
    }
    __syncthreads();

    const unsigned short* Bp = W2b + (size_t)e * (size_t)DIMX * HIDX + (size_t)n0 * HIDX;
    const unsigned short* Ap = Hb + (size_t)(base + mt * BM) * HIDX;

    auto issue = [&](int buf, int k0) {
        unsigned short* dA = smem + buf * (2 * TILE);
        unsigned short* dB = dA + TILE;
        #pragma unroll
        for (int i = tid; i < BM * (BK / 8); i += 256) {
            int r = i >> 3, ch = (i & 7) * 8;
            async_cp16(lds_off(&dA[r * LDA + ch]), &Ap[(size_t)r * HIDX + k0 + ch]);
            async_cp16(lds_off(&dB[r * LDA + ch]), &Bp[(size_t)r * HIDX + k0 + ch]);
        }
    };

    v8f acc[4][2];
    #pragma unroll
    for (int mi = 0; mi < 4; ++mi)
        #pragma unroll
        for (int ni = 0; ni < 2; ++ni)
            acc[mi][ni] = (v8f){0,0,0,0,0,0,0,0};

    issue(0, 0);
    constexpr int NK = HIDX / BK;
    for (int it = 0; it < NK; ++it) {
        int cur = it & 1;
        if (it + 1 < NK) {
            issue(1 - cur, (it + 1) * BK);
            asm volatile("s_wait_asynccnt 8" ::: "memory");
        } else {
            asm volatile("s_wait_asynccnt 0" ::: "memory");
        }
        __syncthreads();

        const unsigned short* cA = smem + cur * (2 * TILE);
        const unsigned short* cB = cA + TILE;

        #pragma unroll
        for (int kk = 0; kk < BK; kk += 32) {
            int kA0 = kk + lhi * 16, kA1 = kk + lhi * 16 + 8;
            int kB0 = kk + lhi * 16;
            v16bf af[4], bf2[2];
            #pragma unroll
            for (int mi = 0; mi < 4; ++mi) {
                int row = wm * 64 + mi * 16 + lrow;
                af[mi] = frag16(&cA[row * LDA + kA0], &cA[row * LDA + kA1]);
            }
            #pragma unroll
            for (int ni = 0; ni < 2; ++ni) {
                int row = wn * 32 + ni * 16 + lrow;
                bf2[ni] = frag16(&cB[row * LDA + kB0], &cB[row * LDA + kB0 + 8]);
            }
            #pragma unroll
            for (int mi = 0; mi < 4; ++mi)
                #pragma unroll
                for (int ni = 0; ni < 2; ++ni)
                    acc[mi][ni] = __builtin_amdgcn_wmma_f32_16x16x32_bf16(
                        false, af[mi], false, bf2[ni], (short)0, acc[mi][ni], false, false);
        }
        __syncthreads();
    }

    #pragma unroll
    for (int mi = 0; mi < 4; ++mi)
        #pragma unroll
        for (int ni = 0; ni < 2; ++ni)
            #pragma unroll
            for (int j = 0; j < 8; ++j) {
                int lm = wm * 64 + mi * 16 + lhi * 8 + j;
                int rT = mt * BM + lm;
                if (rT < c) {
                    int col = n0 + wn * 32 + ni * 16 + lrow;
                    float v = acc[mi][ni][j] * sWgt[lm];
                    unsafeAtomicAdd(&out[(size_t)sTok[lm] * DIMX + col], v);
                }
            }
}

// ---------------- host launcher ----------------
extern "C" void kernel_launch(void* const* d_in, const int* in_sizes, int n_in,
                              void* d_out, int out_size, void* d_ws, size_t ws_size,
                              hipStream_t stream) {
    (void)in_sizes; (void)n_in; (void)out_size; (void)ws_size;
    const float* x  = (const float*)d_in[0];
    const float* Wg = (const float*)d_in[1];
    const float* W1 = (const float*)d_in[2];
    const float* W3 = (const float*)d_in[3];
    const float* W2 = (const float*)d_in[4];
    float* out = (float*)d_out;

    char* p = (char*)d_ws;
    auto alloc = [&](size_t bytes) -> char* {
        char* r = p; p += (bytes + 255) & ~(size_t)255; return r;
    };
    const size_t nW = (size_t)NEXP * HIDX * DIMX;      // elems per weight tensor
    unsigned short* W1b = (unsigned short*)alloc(nW * 2);
    unsigned short* W3b = (unsigned short*)alloc(nW * 2);
    unsigned short* W2b = (unsigned short*)alloc(nW * 2);
    unsigned short* xb  = (unsigned short*)alloc((size_t)TOKS * DIMX * 2);
    unsigned short* Hb  = (unsigned short*)alloc((size_t)(PAIRS + BM) * HIDX * 2);
    int*   ptok = (int*)  alloc((PAIRS + BM) * sizeof(int));
    float* pwgt = (float*)alloc((PAIRS + BM) * sizeof(float));
    int*   cntp = (int*)  alloc(NEXP * sizeof(int));
    int*   curp = (int*)  alloc(NEXP * sizeof(int));
    int*   offp = (int*)  alloc((NEXP + 1) * sizeof(int));
    int*   tkid = (int*)  alloc((size_t)TOKS * 2 * sizeof(int));
    float* tkw  = (float*)alloc((size_t)TOKS * 2 * sizeof(float));

    // allow >64KB dynamic LDS (gfx1250: 320KB per WGP)
    constexpr size_t SH1 = (size_t)2 * 3 * TILE * 2;   // 110592 B
    constexpr size_t SH2 = (size_t)2 * 2 * TILE * 2;   //  73728 B
    (void)hipFuncSetAttribute((const void*)moe_gemm1,
                              hipFuncAttributeMaxDynamicSharedMemorySize, (int)SH1);
    (void)hipFuncSetAttribute((const void*)moe_gemm2,
                              hipFuncAttributeMaxDynamicSharedMemorySize, (int)SH2);

    zero_kernel<<<(TOKS * DIMX) / 256, 256, 0, stream>>>(out, cntp);
    cvt_kernel<<<(int)(nW / 4 / 256), 256, 0, stream>>>(W1, W1b, (int)(nW / 4));
    cvt_kernel<<<(int)(nW / 4 / 256), 256, 0, stream>>>(W3, W3b, (int)(nW / 4));
    cvt_kernel<<<(int)(nW / 4 / 256), 256, 0, stream>>>(W2, W2b, (int)(nW / 4));
    cvt_kernel<<<(TOKS * DIMX / 4) / 256, 256, 0, stream>>>(x, xb, TOKS * DIMX / 4);
    gating_kernel<<<TOKS / 8, 256, 0, stream>>>(x, Wg, tkid, tkw, cntp);
    scan_kernel<<<1, 32, 0, stream>>>(cntp, offp, curp);
    scatter_kernel<<<TOKS / 256, 256, 0, stream>>>(tkid, tkw, curp, ptok, pwgt);

    dim3 g1(NEXP * (TOKS / BM), HIDX / BN);   // 256 x 22
    moe_gemm1<<<g1, 256, SH1, stream>>>(xb, W1b, W3b, Hb, ptok, cntp, offp);
    dim3 g2(NEXP * (TOKS / BM), DIMX / BN);   // 256 x 8
    moe_gemm2<<<g2, 256, SH2, stream>>>(Hb, W2b, out, ptok, pwgt, cntp, offp);
}
